// InteractionNetwork_4861902979311
// MI455X (gfx1250) — matrix operand order repaired and checked by reading, not compiled, not run on hardware
//
#include <hip/hip_runtime.h>
#include <hip/hip_bf16.h>

typedef __attribute__((ext_vector_type(16))) _Float16 v16h;
typedef __attribute__((ext_vector_type(8)))  _Float16 v8h;
typedef __attribute__((ext_vector_type(2)))  _Float16 v2h;
typedef __attribute__((ext_vector_type(8)))  float    v8f;
typedef __attribute__((ext_vector_type(4)))  float    v4f;

#define N_OBJ 2048
#define N_REL 32768
#define D_OBJ 64
#define D_REL 32
#define D_EFF 64
#define H_REL 128
#define H_OBJ 128
#define D_OUT 3
#define K_IN  160           // 2*D_OBJ + D_REL
#define ACT_STRIDE 168      // halfs; 336B row pitch -> staggered LDS banks

// ---------------------------------------------------------------- utilities

__global__ void k_zero(float* __restrict__ p, int n) {
  int i = blockIdx.x * blockDim.x + threadIdx.x;
  if (i < n) p[i] = 0.0f;
}

// Recover integer index from a one-hot row: idx = sum_j j * row[j] (exact in
// f32: exactly one nonzero, j < 2048). One wave per relation row. This kernel
// IS the roofline floor (512 MB @ 23.3 TB/s), so read with b128 vector loads:
// 512 coalesced bytes per wave per iteration, 16 iterations per row.
__global__ void __launch_bounds__(256) k_extract_idx(
    const float* __restrict__ rr, const float* __restrict__ rs,
    int* __restrict__ recv, int* __restrict__ send) {
  int wave = threadIdx.x >> 5;
  int lane = threadIdx.x & 31;
  int rel  = blockIdx.x * 8 + wave;
  const v4f* row = (const v4f*)((blockIdx.y == 0 ? rr : rs) + (size_t)rel * N_OBJ);
  float s = 0.0f;
#pragma unroll 4
  for (int i = 0; i < N_OBJ / 128; ++i) {          // 16 iterations
    v4f v = row[i * 32 + lane];                    // global_load_b128, coalesced
    float j0 = (float)(i * 128 + lane * 4);
    s += v.x * j0 + v.y * (j0 + 1.0f) + v.z * (j0 + 2.0f) + v.w * (j0 + 3.0f);
  }
#pragma unroll
  for (int off = 16; off > 0; off >>= 1) s += __shfl_down(s, off, 32);
  if (lane == 0) (blockIdx.y == 0 ? recv : send)[rel] = (int)(s + 0.5f);
}

// W[K][N] f32 (row-major) -> Wt[Npad][K] f16 (transposed), zero-padded cols.
__global__ void k_wt(const float* __restrict__ W, _Float16* __restrict__ Wt,
                     int K, int N, int Npad) {
  int i = blockIdx.x * blockDim.x + threadIdx.x;
  if (i >= K * Npad) return;
  int k = i % K, n = i / K;
  Wt[(size_t)n * K + k] = (n < N) ? (_Float16)W[(size_t)k * N + n] : (_Float16)0.0f;
}

// ------------------------------------------------------- fused WMMA layer
// actIn/actOut: LDS tiles, row pitch ACT_STRIDE halfs (may alias: A-frags are
// fully register-resident before any store, and each wave owns rows
// [row0,row0+16) exclusively -> no barriers needed).
// Wt: global f16 [N][K]; B frag = one contiguous 32B load per lane.
template<int K, int N, bool RELU>
__device__ __forceinline__ void mlp_layer(
    const _Float16* actIn, _Float16* actOut,
    const _Float16* __restrict__ Wt, const float* __restrict__ bias,
    int lane, int row0) {
  constexpr int KC = K / 32;
  const int h  = lane >> 4;
  const int nl = lane & 15;
  union { v16h v; v8h p[2]; } a[KC];
  const _Float16* ab = actIn + (size_t)(row0 + nl) * ACT_STRIDE;
#pragma unroll
  for (int kc = 0; kc < KC; ++kc) {          // A layout: e0..7 <-> k=16h+0..7,
    a[kc].p[0] = *(const v8h*)(ab + kc * 32 + 16 * h);       // e8..15 <-> k=16h+8..15
    a[kc].p[1] = *(const v8h*)(ab + kc * 32 + 16 * h + 8);
  }
#pragma unroll
  for (int nt = 0; nt < N / 16; ++nt) {
    const int n = nt * 16 + nl;
    const float bv = bias[n];
    v8f c = {bv, bv, bv, bv, bv, bv, bv, bv};           // bias pre-broadcast into C
    const _Float16* wp = Wt + (size_t)n * K + 16 * h;   // B: k = 16h + e (contig 32B)
#pragma unroll
    for (int kc = 0; kc < KC; ++kc) {
      v16h b = *(const v16h*)(wp + kc * 32);
      c = __builtin_amdgcn_wmma_f32_16x16x32_f16(false, a[kc].v, false, b,
                                                 (short)0, c, false, false);
    }
#pragma unroll
    for (int r = 0; r < 8; ++r) {                        // C/D: m = r + 8h, col n
      float v = RELU ? fmaxf(c[r], 0.0f) : c[r];
      actOut[(size_t)(row0 + r + 8 * h) * ACT_STRIDE + n] = (_Float16)v;
    }
  }
}

// --------------------------------------------- relation MLP (4 fused layers)
__global__ void __launch_bounds__(256) k_relation(
    const float* __restrict__ obj, const float* __restrict__ ra,
    const int* __restrict__ recv, const int* __restrict__ send,
    const _Float16* __restrict__ w1t, const _Float16* __restrict__ w2t,
    const _Float16* __restrict__ w3t, const _Float16* __restrict__ w4t,
    const float* __restrict__ b1, const float* __restrict__ b2,
    const float* __restrict__ b3, const float* __restrict__ b4,
    float* __restrict__ e_agg) {
  __shared__ __align__(16) _Float16 act[128 * ACT_STRIDE];
  const int wave = threadIdx.x >> 5;
  const int lane = threadIdx.x & 31;
  const int row0 = wave * 16;
  const int tileBase = blockIdx.x * 128;

  // Stage b = [obj[recv], obj[send], ra] as f16 into this wave's 16-row slab.
  // Per row: lanes 0..15 fetch obj[recv] (16 x b128), lanes 16..31 obj[send],
  // lanes 0..7 fetch ra (8 x b128); f16 pairs stored as ds_store_b32.
  for (int r = 0; r < 16; ++r) {
    const int rel = tileBase + row0 + r;
    _Float16* dst = act + (size_t)(row0 + r) * ACT_STRIDE;
    const int h  = lane >> 4;
    const int sl = lane & 15;
    const float* src = obj + (size_t)(h ? send[rel] : recv[rel]) * D_OBJ;
    v4f v = *(const v4f*)(src + sl * 4);
    v2h p01 = {(_Float16)v.x, (_Float16)v.y};
    v2h p23 = {(_Float16)v.z, (_Float16)v.w};
    *(v2h*)(dst + h * D_OBJ + sl * 4)     = p01;
    *(v2h*)(dst + h * D_OBJ + sl * 4 + 2) = p23;
    if (lane < 8) {
      v4f a = *(const v4f*)(ra + (size_t)rel * D_REL + lane * 4);
      v2h a01 = {(_Float16)a.x, (_Float16)a.y};
      v2h a23 = {(_Float16)a.z, (_Float16)a.w};
      *(v2h*)(dst + 2 * D_OBJ + lane * 4)     = a01;
      *(v2h*)(dst + 2 * D_OBJ + lane * 4 + 2) = a23;
    }
  }

  mlp_layer<K_IN,  H_REL, true>(act, act, w1t, b1, lane, row0);
  mlp_layer<H_REL, H_REL, true>(act, act, w2t, b2, lane, row0);
  mlp_layer<H_REL, H_REL, true>(act, act, w3t, b3, lane, row0);

  // Layer 4 (128->64) + ReLU, fused with receiver scatter-add (rr.T @ e).
  {
    const int h = lane >> 4, nl = lane & 15;
    union { v16h v; v8h p[2]; } a[4];
    const _Float16* ab = act + (size_t)(row0 + nl) * ACT_STRIDE;
#pragma unroll
    for (int kc = 0; kc < 4; ++kc) {
      a[kc].p[0] = *(const v8h*)(ab + kc * 32 + 16 * h);
      a[kc].p[1] = *(const v8h*)(ab + kc * 32 + 16 * h + 8);
    }
#pragma unroll
    for (int nt = 0; nt < D_EFF / 16; ++nt) {
      const int n = nt * 16 + nl;
      const float bv = b4[n];
      v8f c = {bv, bv, bv, bv, bv, bv, bv, bv};
      const _Float16* wp = w4t + (size_t)n * H_REL + 16 * h;
#pragma unroll
      for (int kc = 0; kc < 4; ++kc) {
        v16h b = *(const v16h*)(wp + kc * 32);
        c = __builtin_amdgcn_wmma_f32_16x16x32_f16(false, a[kc].v, false, b,
                                                   (short)0, c, false, false);
      }
#pragma unroll
      for (int r = 0; r < 8; ++r) {
        const float v = fmaxf(c[r], 0.0f);
        const int rel = tileBase + row0 + r + 8 * h;
        atomicAdd(e_agg + (size_t)recv[rel] * D_EFF + n, v);
      }
    }
  }
}

// -------------------------------------------------- object MLP (2 layers)
__global__ void __launch_bounds__(256) k_object(
    const float* __restrict__ obj, const float* __restrict__ e_agg,
    const _Float16* __restrict__ ow1t, const _Float16* __restrict__ ow2t,
    const float* __restrict__ ob1, const float* __restrict__ ob2,
    float* __restrict__ out) {
  __shared__ __align__(16) _Float16 act[128 * ACT_STRIDE];
  const int wave = threadIdx.x >> 5;
  const int lane = threadIdx.x & 31;
  const int row0 = wave * 16;
  const int base = blockIdx.x * 128;

  // Stage c = [obj, e_agg] as f16; lanes 0..15 -> obj, lanes 16..31 -> e_agg.
  for (int r = 0; r < 16; ++r) {
    const int node = base + row0 + r;
    _Float16* dst = act + (size_t)(row0 + r) * ACT_STRIDE;
    const int h  = lane >> 4;
    const int sl = lane & 15;
    const float* src = h ? (e_agg + (size_t)node * D_EFF)
                         : (obj   + (size_t)node * D_OBJ);
    v4f v = *(const v4f*)(src + sl * 4);
    v2h p01 = {(_Float16)v.x, (_Float16)v.y};
    v2h p23 = {(_Float16)v.z, (_Float16)v.w};
    *(v2h*)(dst + h * D_OBJ + sl * 4)     = p01;
    *(v2h*)(dst + h * D_OBJ + sl * 4 + 2) = p23;
  }

  mlp_layer<D_OBJ + D_EFF, H_OBJ, true>(act, act, ow1t, ob1, lane, row0);

  // Final 128->3 projection (N padded to 16 with zero weight cols).
  const int h = lane >> 4, nl = lane & 15;
  union { v16h v; v8h p[2]; } a[4];
  const _Float16* ab = act + (size_t)(row0 + nl) * ACT_STRIDE;
#pragma unroll
  for (int kc = 0; kc < 4; ++kc) {
    a[kc].p[0] = *(const v8h*)(ab + kc * 32 + 16 * h);
    a[kc].p[1] = *(const v8h*)(ab + kc * 32 + 16 * h + 8);
  }
  const float bv = (nl < D_OUT) ? ob2[nl] : 0.0f;
  v8f c = {bv, bv, bv, bv, bv, bv, bv, bv};
  const _Float16* wp = ow2t + (size_t)nl * H_OBJ + 16 * h;
#pragma unroll
  for (int kc = 0; kc < 4; ++kc) {
    v16h b = *(const v16h*)(wp + kc * 32);
    c = __builtin_amdgcn_wmma_f32_16x16x32_f16(false, a[kc].v, false, b,
                                               (short)0, c, false, false);
  }
  if (nl < D_OUT) {
#pragma unroll
    for (int r = 0; r < 8; ++r)
      out[(size_t)(base + row0 + r + 8 * h) * D_OUT + nl] = c[r];
  }
}

// ---------------------------------------------------------------- launcher
extern "C" void kernel_launch(void* const* d_in, const int* in_sizes, int n_in,
                              void* d_out, int out_size, void* d_ws, size_t ws_size,
                              hipStream_t stream) {
  (void)in_sizes; (void)n_in; (void)out_size; (void)ws_size;
  const float* obj   = (const float*)d_in[0];
  const float* rr    = (const float*)d_in[1];
  const float* rs    = (const float*)d_in[2];
  const float* ra    = (const float*)d_in[3];
  // d_in[4] = x (zero-width), unused
  const float* rm_w1 = (const float*)d_in[5];  const float* rm_b1 = (const float*)d_in[6];
  const float* rm_w2 = (const float*)d_in[7];  const float* rm_b2 = (const float*)d_in[8];
  const float* rm_w3 = (const float*)d_in[9];  const float* rm_b3 = (const float*)d_in[10];
  const float* rm_w4 = (const float*)d_in[11]; const float* rm_b4 = (const float*)d_in[12];
  const float* om_w1 = (const float*)d_in[13]; const float* om_b1 = (const float*)d_in[14];
  const float* om_w2 = (const float*)d_in[15]; const float* om_b2 = (const float*)d_in[16];
  float* out = (float*)d_out;

  char* ws = (char*)d_ws;
  size_t off = 0;
  int*      recv  = (int*)(ws + off);      off += (size_t)N_REL * 4;
  int*      send  = (int*)(ws + off);      off += (size_t)N_REL * 4;
  float*    e_agg = (float*)(ws + off);    off += (size_t)N_OBJ * D_EFF * 4;
  _Float16* w1t   = (_Float16*)(ws + off); off += (size_t)H_REL * K_IN * 2;
  _Float16* w2t   = (_Float16*)(ws + off); off += (size_t)H_REL * H_REL * 2;
  _Float16* w3t   = (_Float16*)(ws + off); off += (size_t)H_REL * H_REL * 2;
  _Float16* w4t   = (_Float16*)(ws + off); off += (size_t)D_EFF * H_REL * 2;
  _Float16* ow1t  = (_Float16*)(ws + off); off += (size_t)H_OBJ * H_OBJ * 2;
  _Float16* ow2t  = (_Float16*)(ws + off); off += (size_t)16 * H_OBJ * 2;

  // 1) zero the aggregation buffer
  k_zero<<<(N_OBJ * D_EFF + 255) / 256, 256, 0, stream>>>(e_agg, N_OBJ * D_EFF);
  // 2) recover indices from the one-hot matrices (the 512MB bandwidth floor)
  k_extract_idx<<<dim3(N_REL / 8, 2), 256, 0, stream>>>(rr, rs, recv, send);
  // 3) f16 transposed weights ([N][K]) for direct B-fragment loads
  k_wt<<<(K_IN * H_REL + 255) / 256, 256, 0, stream>>>(rm_w1, w1t, K_IN, H_REL, H_REL);
  k_wt<<<(H_REL * H_REL + 255) / 256, 256, 0, stream>>>(rm_w2, w2t, H_REL, H_REL, H_REL);
  k_wt<<<(H_REL * H_REL + 255) / 256, 256, 0, stream>>>(rm_w3, w3t, H_REL, H_REL, H_REL);
  k_wt<<<(H_REL * D_EFF + 255) / 256, 256, 0, stream>>>(rm_w4, w4t, H_REL, D_EFF, D_EFF);
  k_wt<<<(H_OBJ * H_OBJ + 255) / 256, 256, 0, stream>>>(om_w1, ow1t, H_OBJ, H_OBJ, H_OBJ);
  k_wt<<<(H_OBJ * 16 + 255) / 256, 256, 0, stream>>>(om_w2, ow2t, H_OBJ, D_OUT, 16);
  // 4) fused 4-layer relation MLP + receiver scatter-add
  k_relation<<<N_REL / 128, 256, 0, stream>>>(obj, ra, recv, send,
                                              w1t, w2t, w3t, w4t,
                                              rm_b1, rm_b2, rm_b3, rm_b4, e_agg);
  // 5) fused 2-layer object MLP -> p[2048][3]
  k_object<<<N_OBJ / 128, 256, 0, stream>>>(obj, e_agg, ow1t, ow2t, om_b1, om_b2, out);
}
